// UniEA_69166153335082
// MI455X (gfx1250) — matrix operand order repaired and checked by reading, not compiled, not run on hardware
//
#include <hip/hip_runtime.h>
#include <math.h>

typedef __attribute__((ext_vector_type(16))) _Float16 v16h;
typedef __attribute__((ext_vector_type(4)))  _Float16 v4h;
typedef __attribute__((ext_vector_type(2)))  _Float16 v2h;
typedef __attribute__((ext_vector_type(2)))  __fp16   v2fp16;
typedef __attribute__((ext_vector_type(8)))  float    v8f;
typedef __attribute__((ext_vector_type(4)))  float    vf4;
typedef __attribute__((ext_vector_type(2)))  float    vf2;

static constexpr int   kN   = 6000;
static constexpr int   kD   = 128;
static constexpr int   kR   = 1000;
static constexpr int   kHD  = 256;   // H*D
static constexpr float kMAXN = 1.0f - 4e-3f;

union AFrag { v16h v; v2h p[8]; _Float16 e[16]; };
union H4    { v4h v; v2h p[2]; };

// pack two f32 -> two f16 (RTZ), reconciling __fp16/_Float16 vector types
static __device__ __forceinline__ v2h pk2h(float x, float y) {
  union { v2fp16 a; v2h b; } u;
  u.a = __builtin_amdgcn_cvt_pkrtz(x, y);
  return u.b;
}

// ---------------------------------------------------------------------------
// Tiled WMMA GEMM: C[M,Nc] = A[M,K] * B[K,Nc]  (fp32 in/out, f16 MACs).
// REQUIRES M%16==0, Nc%16==0, lda%4==0, (transB? ldb%4==0 : 1) -- true at all
// call sites. K tail handled. One wave per 16x16 C tile, 4 waves/block,
// 2-chunk K unroll with dual accumulators. A staged [row][k], B staged
// transposed [col][k] (pad 36 halves = 72B row stride) so fragment gathers
// are packed 32-bit DS loads of f16 pairs.
// ---------------------------------------------------------------------------
__global__ __launch_bounds__(128)
void k_gemm(const float* __restrict__ A, const float* __restrict__ B,
            float* __restrict__ C, int M, int Nc, int K,
            int lda, int ldb, int ldc, int transB) {
  __shared__ _Float16 sA[4][2][16][36];
  __shared__ _Float16 sBt[4][2][16][36];
  const int w = threadIdx.x >> 5, lane = threadIdx.x & 31;
  const int tilesN = Nc >> 4;
  const int tile = blockIdx.x * 4 + w;
  if (tile >= (M >> 4) * tilesN) return;
  const int m0 = (tile / tilesN) * 16;
  const int n0 = (tile % tilesN) * 16;
  const int nl = lane & 15, half = lane >> 4;
  const int rrow = lane >> 1, rch = (lane & 1) * 16;   // row-half staging map

  // stage 16 consecutive floats of one row into dst[rrow][rch..rch+15]
  auto stageRow = [&](_Float16* dst, const float* p) {
    vf4 f0 = *(const vf4*)(p);
    vf4 f1 = *(const vf4*)(p + 4);
    vf4 f2 = *(const vf4*)(p + 8);
    vf4 f3 = *(const vf4*)(p + 12);
    H4 h0, h1, h2, h3;
    h0.p[0] = pk2h(f0.x, f0.y); h0.p[1] = pk2h(f0.z, f0.w);
    h1.p[0] = pk2h(f1.x, f1.y); h1.p[1] = pk2h(f1.z, f1.w);
    h2.p[0] = pk2h(f2.x, f2.y); h2.p[1] = pk2h(f2.z, f2.w);
    h3.p[0] = pk2h(f3.x, f3.y); h3.p[1] = pk2h(f3.z, f3.w);
    *(v4h*)(dst)      = h0.v;
    *(v4h*)(dst + 4)  = h1.v;
    *(v4h*)(dst + 8)  = h2.v;
    *(v4h*)(dst + 12) = h3.v;
  };
  auto stageA = [&](int cb, int kk) {
    stageRow(&sA[w][cb][rrow][rch], A + (size_t)(m0 + rrow) * lda + kk + rch);
  };
  auto stageB = [&](int cb, int kk) {
    if (transB) {   // B is [Nc,K]: K contiguous per row -> b128 loads
      stageRow(&sBt[w][cb][rrow][rch], B + (size_t)(n0 + rrow) * ldb + kk + rch);
    } else {        // B is [K,Nc]: strided column loads, split load/store
      const float* p = B + (size_t)kk * ldb + n0 + nl;
      float fv[16];
      #pragma unroll
      for (int i = 0; i < 8; ++i) {
        int r = 2 * (2 * i + half);
        fv[2 * i]     = p[(size_t)r * ldb];
        fv[2 * i + 1] = p[(size_t)(r + 1) * ldb];
      }
      #pragma unroll
      for (int i = 0; i < 8; ++i) {
        int r = 2 * (2 * i + half);
        *(v2h*)&sBt[w][cb][nl][r] = pk2h(fv[2 * i], fv[2 * i + 1]);
      }
    }
  };
  auto domma = [&](int cb, v8f acc) -> v8f {
    AFrag a, b;
    #pragma unroll
    for (int vv = 0; vv < 8; ++vv) {
      int kb  = (vv >> 2) * 16 + half * 8 + (vv & 3) * 2;
      int kb2 = (vv >> 2) * 16 + half * 8 + (vv & 3) * 2;
      a.p[vv] = *(const v2h*)&sA[w][cb][nl][kb];
      b.p[vv] = *(const v2h*)&sBt[w][cb][nl][kb2];
    }
    return __builtin_amdgcn_wmma_f32_16x16x32_f16(false, a.v, false, b.v,
                                                  (short)0, acc, false, false);
  };

  v8f acc0 = {}, acc1 = {};
  const int kmain = K & ~31;
  int k0 = 0;
  for (; k0 + 64 <= kmain; k0 += 64) {
    __builtin_prefetch(A + (size_t)(m0 + rrow) * lda + k0 + 64, 0, 1);
    stageA(0, k0);      stageB(0, k0);
    stageA(1, k0 + 32); stageB(1, k0 + 32);
    acc0 = domma(0, acc0);
    acc1 = domma(1, acc1);
  }
  if (k0 < kmain) {
    stageA(0, k0); stageB(0, k0);
    acc0 = domma(0, acc0);
    k0 += 32;
  }
  if (k0 < K) {  // checked K tail, zero-filled
    #pragma unroll 4
    for (int i = 0; i < 16; ++i)
      sA[w][0][i][lane] =
          (_Float16)((k0 + lane < K) ? A[(size_t)(m0 + i) * lda + k0 + lane] : 0.f);
    for (int r = half; r < 32; r += 2) {
      int gk = k0 + r;
      float v = 0.f;
      if (gk < K)
        v = transB ? B[(size_t)(n0 + nl) * ldb + gk] : B[(size_t)gk * ldb + n0 + nl];
      sBt[w][0][nl][r] = (_Float16)v;
    }
    acc0 = domma(0, acc0);
  }
  v8f acc = acc0 + acc1;
  float* pC = C + (size_t)(m0 + half * 8) * ldc + n0 + nl;
  #pragma unroll
  for (int r = 0; r < 8; ++r) pC[(size_t)r * ldc] = acc[r];
}

// ---------------------------------------------------------------------------
// GAT fused attention pass 1: per-row softmax stats (max, sum of exp)
// ---------------------------------------------------------------------------
__global__ __launch_bounds__(256)
void k_gat_stats(const float* __restrict__ es, const float* __restrict__ ed,
                 const float* __restrict__ adj, float* rmax, float* rsum, int n_) {
  __shared__ float red[256];
  const int n = blockIdx.x, t = threadIdx.x;
  const float esn = es[n];
  float mx = -1e30f;
  for (int m = t; m < n_; m += 256)
    if (adj[(size_t)n * n_ + m] > 0.f) {
      float e = esn + ed[m]; e = e > 0.f ? e : 0.2f * e;
      mx = fmaxf(mx, e);
    }
  red[t] = mx; __syncthreads();
  for (int s = 128; s > 0; s >>= 1) { if (t < s) red[t] = fmaxf(red[t], red[t + s]); __syncthreads(); }
  mx = red[0]; __syncthreads();
  float sm = 0.f;
  for (int m = t; m < n_; m += 256)
    if (adj[(size_t)n * n_ + m] > 0.f) {
      float e = esn + ed[m]; e = e > 0.f ? e : 0.2f * e;
      sm += __expf(e - mx);
    }
  red[t] = sm; __syncthreads();
  for (int s = 128; s > 0; s >>= 1) { if (t < s) red[t] += red[t + s]; __syncthreads(); }
  if (t == 0) { rmax[n] = mx; rsum[n] = fmaxf(red[0], 1e-30f); }
}

// ---------------------------------------------------------------------------
// GAT fused attention pass 2: out = softmax(mask(lrelu(es_n+ed_m))) @ h.
// Attention A-tile built in registers (batched float2 adj/ed loads, pk pack);
// h B-tile staged transposed in LDS. REQUIRES n_%16==0, d_%16==0.
// ---------------------------------------------------------------------------
__global__ __launch_bounds__(128)
void k_gat_agg(const float* __restrict__ h, const float* __restrict__ es,
               const float* __restrict__ ed, const float* __restrict__ rmax,
               const float* __restrict__ rsum, const float* __restrict__ adj,
               float* __restrict__ out, int n_, int d_) {
  __shared__ _Float16 sBt[4][16][36];
  const int w = threadIdx.x >> 5, lane = threadIdx.x & 31;
  const int tilesC = d_ >> 4;
  const int tile = blockIdx.x * 4 + w;
  if (tile >= (n_ >> 4) * tilesC) return;
  const int n0 = (tile / tilesC) * 16;
  const int d0 = (tile % tilesC) * 16;
  const int nl = lane & 15, half = lane >> 4;
  const int n = n0 + nl;
  const float esn = es[n];
  const float mxn = rmax[n];
  const float isn = 1.f / rsum[n];
  const float* adjrow = adj + (size_t)n * n_;

  v8f acc = {};
  const int mmain = n_ & ~31;
  int m0 = 0;
  for (; m0 < mmain; m0 += 32) {
    // stage h tile transposed: split load / store phases
    const float* p = h + (size_t)m0 * d_ + d0 + nl;
    float fv[16];
    #pragma unroll
    for (int i = 0; i < 8; ++i) {
      int r = 2 * (2 * i + half);
      fv[2 * i]     = p[(size_t)r * d_];
      fv[2 * i + 1] = p[(size_t)(r + 1) * d_];
    }
    #pragma unroll
    for (int i = 0; i < 8; ++i) {
      int r = 2 * (2 * i + half);
      *(v2h*)&sBt[w][nl][r] = pk2h(fv[2 * i], fv[2 * i + 1]);
    }
    // batched adj/ed pair loads, then attention math, then pack
    vf2 ad[8], ev[8];
    #pragma unroll
    for (int vv = 0; vv < 8; ++vv) {
      int m = m0 + (vv >> 2) * 16 + half * 8 + (vv & 3) * 2;
      ad[vv] = *(const vf2*)&adjrow[m];
      ev[vv] = *(const vf2*)&ed[m];
    }
    AFrag a, b;
    #pragma unroll
    for (int vv = 0; vv < 8; ++vv) {
      float a0 = 0.f, a1 = 0.f;
      if (ad[vv].x > 0.f) { float e2 = esn + ev[vv].x; e2 = e2 > 0.f ? e2 : 0.2f * e2;
                            a0 = __expf(e2 - mxn) * isn; }
      if (ad[vv].y > 0.f) { float e2 = esn + ev[vv].y; e2 = e2 > 0.f ? e2 : 0.2f * e2;
                            a1 = __expf(e2 - mxn) * isn; }
      a.p[vv] = pk2h(a0, a1);
      b.p[vv] = *(const v2h*)&sBt[w][nl][(vv >> 2) * 16 + half * 8 + (vv & 3) * 2];
    }
    acc = __builtin_amdgcn_wmma_f32_16x16x32_f16(false, a.v, false, b.v,
                                                 (short)0, acc, false, false);
  }
  if (m0 < n_) {  // checked m tail
    for (int r = half; r < 32; r += 2) {
      int gm = m0 + r;
      sBt[w][nl][r] = (_Float16)((gm < n_) ? h[(size_t)gm * d_ + d0 + nl] : 0.f);
    }
    AFrag a, b;
    #pragma unroll
    for (int vv = 0; vv < 8; ++vv) {
      int kb = (vv >> 2) * 16 + half * 8 + (vv & 3) * 2;
      #pragma unroll
      for (int j = 0; j < 2; ++j) {
        int m = m0 + kb + j;
        float att = 0.f;
        if (m < n_ && adjrow[m] > 0.f) {
          float e2 = esn + ed[m]; e2 = e2 > 0.f ? e2 : 0.2f * e2;
          att = __expf(e2 - mxn) * isn;
        }
        a.e[2 * vv + j] = (_Float16)att;
      }
      b.p[vv] = *(const v2h*)&sBt[w][nl][(vv >> 2) * 16 + half * 8 + (vv & 3) * 2];
    }
    acc = __builtin_amdgcn_wmma_f32_16x16x32_f16(false, a.v, false, b.v,
                                                 (short)0, acc, false, false);
  }
  float* pO = out + (size_t)(n0 + half * 8) * d_ + d0 + nl;
  #pragma unroll
  for (int r = 0; r < 8; ++r) pO[(size_t)r * d_] = acc[r];
}

// ---------------------------------------------------------------------------
// Row-wise elementwise kernels: one 128-thread block per row (D=128)
// ---------------------------------------------------------------------------
__device__ __forceinline__ float blk_reduce(float v, float* red) {
  const int t = threadIdx.x;
  red[t] = v; __syncthreads();
  for (int s = 64; s > 0; s >>= 1) { if (t < s) red[t] += red[t + s]; __syncthreads(); }
  float r = red[0]; __syncthreads();
  return r;
}

__global__ __launch_bounds__(128)
void k_logmap0(const float* __restrict__ x, float* __restrict__ out,
               int ldx, int ldo, int do_relu) {
  __shared__ float red[128];
  const int n = blockIdx.x, t = threadIdx.x;
  float v = x[(size_t)n * ldx + t];
  float nrm = fmaxf(sqrtf(blk_reduce(v * v, red)), 1e-15f);
  float r = atanhf(fminf(nrm, 1.f - 1e-7f)) / nrm * v;
  if (do_relu) r = fmaxf(r, 0.f);
  out[(size_t)n * ldo + t] = r;
}

__global__ __launch_bounds__(128)
void k_expmap0(const float* __restrict__ u, float* __restrict__ out,
               int ldu, int ldo) {
  __shared__ float red[128];
  const int n = blockIdx.x, t = threadIdx.x;
  float v = u[(size_t)n * ldu + t];
  float nrm = fmaxf(sqrtf(blk_reduce(v * v, red)), 1e-15f);
  float th = tanhf(nrm);
  float s = (th > kMAXN) ? (kMAXN / nrm) : (th / nrm);
  out[(size_t)n * ldo + t] = s * v;
}

__global__ __launch_bounds__(128)
void k_mobius_add_proj(const float* __restrict__ xm, const float* __restrict__ yb,
                       float* __restrict__ out) {
  __shared__ float red[128];
  const int n = blockIdx.x, t = threadIdx.x;
  float x = xm[(size_t)n * kD + t];
  float y = yb[t];
  float x2 = blk_reduce(x * x, red);
  float y2 = blk_reduce(y * y, red);
  float xy = blk_reduce(x * y, red);
  float num = (1.f + 2.f * xy + y2) * x + (1.f - x2) * y;
  float den = fmaxf(1.f + 2.f * xy + x2 * y2, 1e-15f);
  float r = num / den;
  float nrm = fmaxf(sqrtf(blk_reduce(r * r, red)), 1e-15f);
  if (nrm > kMAXN) r = r / nrm * kMAXN;
  out[(size_t)n * kD + t] = r;
}

__global__ __launch_bounds__(128)
void k_gat_dots(const float* __restrict__ h, const float* __restrict__ a_src,
                const float* __restrict__ a_dst, float* es, float* ed) {
  __shared__ float red[128];
  const int n = blockIdx.x, t = threadIdx.x;
  float hv = h[(size_t)n * kD + t];
  float s0 = blk_reduce(hv * a_src[t], red);
  float s1 = blk_reduce(hv * a_dst[t], red);
  if (t == 0) { es[n] = s0; ed[n] = s1; }
}

__global__ __launch_bounds__(128)
void k_gat_combine(const float* __restrict__ o0, const float* __restrict__ o1,
                   float* __restrict__ out, int ldo) {
  __shared__ float red[128];
  const int n = blockIdx.x, t = threadIdx.x;
  float v = 0.5f * (o0[(size_t)n * kD + t] + o1[(size_t)n * kD + t]);
  v = v > 0.f ? v : expm1f(v);                 // ELU
  float nrm = fmaxf(sqrtf(blk_reduce(v * v, red)), 1e-12f);
  out[(size_t)n * ldo + t] = v / nrm;
}

__global__ __launch_bounds__(128)
void k_copy128(const float* __restrict__ src, float* __restrict__ dst,
               int lds_, int ldd) {
  dst[(size_t)blockIdx.x * ldd + threadIdx.x] =
      src[(size_t)blockIdx.x * lds_ + threadIdx.x];
}

// ---------------------------------------------------------------------------
// MHA small kernels
// ---------------------------------------------------------------------------
__global__ void k_mha_attn(const float* __restrict__ q, const float* __restrict__ k,
                           const float* __restrict__ v, float* __restrict__ o, int n_) {
  int idx = blockIdx.x * blockDim.x + threadIdx.x;
  if (idx >= n_ * 2 * 3) return;
  int l = idx % 3, hh = (idx / 3) % 2, n = idx / 6;
  const float* qn = q + (size_t)n * 768 + l * kHD + hh * kD;
  float s[3];
  #pragma unroll
  for (int m = 0; m < 3; ++m) {
    const float* kn = k + (size_t)n * 768 + m * kHD + hh * kD;
    float d = 0.f;
    for (int e = 0; e < kD; ++e) d += qn[e] * kn[e];
    s[m] = d * 0.08838834764831845f;  // 1/sqrt(128)
  }
  float mx = fmaxf(s[0], fmaxf(s[1], s[2]));
  float e0 = __expf(s[0] - mx), e1 = __expf(s[1] - mx), e2 = __expf(s[2] - mx);
  float is = 1.f / (e0 + e1 + e2);
  e0 *= is; e1 *= is; e2 *= is;
  float* on = o + (size_t)n * 768 + l * kHD + hh * kD;
  const float* v0 = v + (size_t)n * 768 + 0 * kHD + hh * kD;
  const float* v1 = v + (size_t)n * 768 + 1 * kHD + hh * kD;
  const float* v2 = v + (size_t)n * 768 + 2 * kHD + hh * kD;
  for (int e = 0; e < kD; ++e) on[e] = e0 * v0[e] + e1 * v1[e] + e2 * v2[e];
}

__global__ __launch_bounds__(128)
void k_mha_ln(const float* __restrict__ fco, const float* __restrict__ seq,
              const float* __restrict__ g, const float* __restrict__ b,
              float* __restrict__ out) {
  __shared__ float red[128];
  const int row = blockIdx.x, t = threadIdx.x;   // row = n*3 + l
  float o = fco[(size_t)row * kD + t] + seq[(size_t)row * kD + t];
  float mu = blk_reduce(o, red) * (1.f / kD);
  float d = o - mu;
  float var = blk_reduce(d * d, red) * (1.f / kD);
  out[(size_t)row * kD + t] = g[t] * d * rsqrtf(var + 1e-6f) + b[t];
}

__global__ __launch_bounds__(128)
void k_mean3(const float* __restrict__ x, float* __restrict__ out) {
  const int n = blockIdx.x, t = threadIdx.x;
  const float* r = x + (size_t)n * 384;
  out[(size_t)n * kD + t] = (r[t] + r[kD + t] + r[2 * kD + t]) * (1.f / 3.f);
}

// ---------------------------------------------------------------------------
// Relation aggregation helpers
// ---------------------------------------------------------------------------
__global__ __launch_bounds__(256)
void k_rowsum(const float* __restrict__ a, float* __restrict__ out, int cols) {
  __shared__ float red[256];
  const int n = blockIdx.x, t = threadIdx.x;
  float s = 0.f;
  for (int c = t; c < cols; c += 256) s += a[(size_t)n * cols + c];
  red[t] = s; __syncthreads();
  for (int st = 128; st > 0; st >>= 1) { if (t < st) red[t] += red[t + st]; __syncthreads(); }
  if (t == 0) out[n] = red[0];
}

__global__ __launch_bounds__(128)
void k_divrow(float* __restrict__ x, const float* __restrict__ s) {
  x[(size_t)blockIdx.x * kD + threadIdx.x] /= s[blockIdx.x];
}

// d_out row: [left(+base) | right], row stride 256
__global__ __launch_bounds__(256)
void k_concat(float* __restrict__ dst, const float* __restrict__ left,
              const float* __restrict__ right, const float* __restrict__ base) {
  const int n = blockIdx.x, t = threadIdx.x;
  float v;
  if (t < kD) {
    v = left[(size_t)n * kD + t];
    if (base) v += base[(size_t)n * kD + t];
  } else {
    v = right[(size_t)n * kD + (t - kD)];
  }
  dst[(size_t)n * 2 * kD + t] = v;
}

// ---------------------------------------------------------------------------
extern "C" void kernel_launch(void* const* d_in, const int* in_sizes, int n_in,
                              void* d_out, int out_size, void* d_ws, size_t ws_size,
                              hipStream_t stream) {
  (void)in_sizes; (void)n_in; (void)out_size; (void)ws_size;
  const float* ent_sr   = (const float*)d_in[0];
  const float* ent_tg   = (const float*)d_in[1];
  const float* rel_sr   = (const float*)d_in[2];
  const float* rel_tg   = (const float*)d_in[3];
  const float* adj_sr   = (const float*)d_in[4];
  const float* adj_tg   = (const float*)d_in[5];
  const float* radj_sr  = (const float*)d_in[6];
  const float* radj_tg  = (const float*)d_in[7];
  const float* gat_W    = (const float*)d_in[8];   // [2,2,128,128]
  const float* gat_asrc = (const float*)d_in[9];   // [2,2,128]
  const float* gat_adst = (const float*)d_in[10];
  const float* Wq       = (const float*)d_in[11];  // [128,256]
  const float* Wk       = (const float*)d_in[12];
  const float* Wv       = (const float*)d_in[13];
  const float* Wfc      = (const float*)d_in[14];  // [256,128]
  const float* ln_g     = (const float*)d_in[15];
  const float* ln_b     = (const float*)d_in[16];
  const float* hgc_W    = (const float*)d_in[17];  // [2,128,128]
  const float* hgc_b    = (const float*)d_in[18];  // [2,128]
  float* out = (float*)d_out;

  float* w = (float*)d_ws;
  auto alloc = [&](size_t n) { float* p = w; w += n; return p; };
  const size_t ND = (size_t)kN * kD;
  float* seq  = alloc(3 * ND);
  float* H0   = alloc(ND);  float* H1 = alloc(ND);
  float* O0   = alloc(ND);  float* O1 = alloc(ND);
  float* es0  = alloc(kN);  float* ed0 = alloc(kN);
  float* es1  = alloc(kN);  float* ed1 = alloc(kN);
  float* rmx0 = alloc(kN);  float* rsm0 = alloc(kN);
  float* rmx1 = alloc(kN);  float* rsm1 = alloc(kN);
  float* qb   = alloc(3 * (size_t)kN * kHD);
  float* kb   = alloc(3 * (size_t)kN * kHD);
  float* vb   = alloc(3 * (size_t)kN * kHD);
  float* ob   = alloc(3 * (size_t)kN * kHD);
  float* fco  = alloc(3 * ND);
  float* lnb  = alloc(3 * ND);
  float* bvec = alloc(ND);
  float* relS = alloc(ND);  float* relT = alloc(ND);
  float* rsR  = alloc(kN);
  float* t0 = alloc(ND); float* t1 = alloc(ND);
  float* t2 = alloc(ND); float* t3 = alloc(ND);
  float* hb = alloc(kD);

  auto gemm = [&](const float* A, int lda, const float* B, int ldb,
                  float* C, int ldc, int M, int Nc, int K, int transB) {
    int tiles = (M / 16) * (Nc / 16);
    k_gemm<<<dim3((tiles + 3) / 4), dim3(128), 0, stream>>>(
        A, B, C, M, Nc, K, lda, ldb, ldc, transB);
  };

  // ---- relation aggregators ----
  gemm(radj_sr, kR, rel_sr, kD, relS, kD, kN, kD, kR, 0);
  k_rowsum<<<kN, 256, 0, stream>>>(radj_sr, rsR, kR);
  k_divrow<<<kN, 128, 0, stream>>>(relS, rsR);
  gemm(radj_tg, kR, rel_tg, kD, relT, kD, kN, kD, kR, 0);
  k_rowsum<<<kN, 256, 0, stream>>>(radj_tg, rsR, kR);
  k_divrow<<<kN, 128, 0, stream>>>(relT, rsR);

  // ---- GAT + multi-range MHA branch ----
  auto branch = [&](const float* x, const float* adj, float* res) {
    k_copy128<<<kN, 128, 0, stream>>>(x, seq, kD, 3 * kD);
    for (int l = 0; l < 2; ++l) {
      const float* xin = seq + (size_t)l * kD;           // row stride 384
      float* Hh[2] = {H0, H1};
      float* Oh[2] = {O0, O1};
      float* esA[2] = {es0, es1}; float* edA[2] = {ed0, ed1};
      float* mxA[2] = {rmx0, rmx1}; float* smA[2] = {rsm0, rsm1};
      for (int hh = 0; hh < 2; ++hh) {
        const float* Wl = gat_W + (size_t)(l * 2 + hh) * kD * kD;
        gemm(xin, 3 * kD, Wl, kD, Hh[hh], kD, kN, kD, kD, 0);
        k_gat_dots<<<kN, 128, 0, stream>>>(Hh[hh],
            gat_asrc + (size_t)(l * 2 + hh) * kD,
            gat_adst + (size_t)(l * 2 + hh) * kD, esA[hh], edA[hh]);
        k_gat_stats<<<kN, 256, 0, stream>>>(esA[hh], edA[hh], adj,
                                            mxA[hh], smA[hh], kN);
        int tiles = (kN / 16) * (kD / 16);
        k_gat_agg<<<dim3((tiles + 3) / 4), dim3(128), 0, stream>>>(
            Hh[hh], esA[hh], edA[hh], mxA[hh], smA[hh], adj, Oh[hh], kN, kD);
      }
      k_gat_combine<<<kN, 128, 0, stream>>>(O0, O1,
          seq + (size_t)(l + 1) * kD, 3 * kD);
    }
    // MHA over seq [N,3,128] flattened [3N,128]
    const int M3 = 3 * kN;
    gemm(seq, kD, Wq, kHD, qb, kHD, M3, kHD, kD, 0);
    gemm(seq, kD, Wk, kHD, kb, kHD, M3, kHD, kD, 0);
    gemm(seq, kD, Wv, kHD, vb, kHD, M3, kHD, kD, 0);
    k_mha_attn<<<(kN * 6 + 255) / 256, 256, 0, stream>>>(qb, kb, vb, ob, kN);
    gemm(ob, kHD, Wfc, kD, fco, kD, M3, kD, kHD, 0);
    k_mha_ln<<<M3, 128, 0, stream>>>(fco, seq, ln_g, ln_b, lnb);
    k_mean3<<<kN, 128, 0, stream>>>(lnb, res);
  };

  // ---- hyperbolic encoder: result (logmap0 of final) left in t1 ----
  auto encode = [&](const float* x, const float* adj) {
    k_expmap0<<<kN, 128, 0, stream>>>(x, t0, kD, kD);
    for (int i = 0; i < 2; ++i) {
      const float* Wl = hgc_W + (size_t)i * kD * kD;
      k_logmap0<<<kN, 128, 0, stream>>>(t0, t1, kD, kD, 0);
      gemm(t1, kD, Wl, kD, t2, kD, kN, kD, kD, 1);        // x @ W.T
      k_expmap0<<<kN, 128, 0, stream>>>(t2, t2, kD, kD);  // mv
      k_expmap0<<<1, 128, 0, stream>>>(hgc_b + (size_t)i * kD, hb, kD, kD);
      k_mobius_add_proj<<<kN, 128, 0, stream>>>(t2, hb, t3);
      k_logmap0<<<kN, 128, 0, stream>>>(t3, t1, kD, kD, 0);
      gemm(adj, kN, t1, kD, t2, kD, kN, kD, kN, 0);       // adj @ logmap0(h)
      k_expmap0<<<kN, 128, 0, stream>>>(t2, t3, kD, kD);
      k_logmap0<<<kN, 128, 0, stream>>>(t3, t1, kD, kD, 1);  // relu in tangent
      k_expmap0<<<kN, 128, 0, stream>>>(t1, t0, kD, kD);
    }
    k_logmap0<<<kN, 128, 0, stream>>>(t0, t1, kD, kD, 0);
  };

  const size_t OS = (size_t)kN * 2 * kD;
  branch(ent_sr, adj_sr, bvec);
  k_concat<<<kN, 256, 0, stream>>>(out + 0 * OS, bvec, relS, nullptr);
  branch(ent_tg, adj_tg, bvec);
  k_concat<<<kN, 256, 0, stream>>>(out + 1 * OS, bvec, relT, nullptr);
  encode(ent_sr, adj_sr);
  k_concat<<<kN, 256, 0, stream>>>(out + 2 * OS, t1, relS, ent_sr);
  encode(ent_tg, adj_tg);
  k_concat<<<kN, 256, 0, stream>>>(out + 3 * OS, t1, relT, ent_tg);
}
